// TransformerBlock_18433999634461
// MI455X (gfx1250) — matrix operand order repaired and checked by reading, not compiled, not run on hardware
//
#include <hip/hip_runtime.h>

// ---------------------------------------------------------------------------
// MI455X / gfx1250 transformer block: bf16 WMMA GEMMs + flash attention.
// Async global->LDS double-buffered GEMM staging (gfx1250 ASYNCcnt path).
// ---------------------------------------------------------------------------

typedef __attribute__((ext_vector_type(16))) __bf16 v16bf;
typedef __attribute__((ext_vector_type(8)))  float  v8f;
typedef __attribute__((ext_vector_type(4)))  unsigned int u32x4;
typedef __attribute__((ext_vector_type(2)))  unsigned int u32x2;
typedef __attribute__((ext_vector_type(4)))  float  f32x4;

struct Frag32B { u32x4 a, b; };

static __device__ __forceinline__ v16bf mkfrag(u32x4 lo, u32x4 hi) {
  Frag32B p{lo, hi};
  return __builtin_bit_cast(v16bf, p);
}

static __device__ __forceinline__ v8f wmma_bf16(v16bf a, v16bf b, v8f c) {
  // D = A(16x32 bf16) * B(32x16 bf16) + C(16x16 f32)
  return __builtin_amdgcn_wmma_f32_16x16x32_bf16(false, a, false, b, (short)0, c,
                                                 false, false);
}

static __device__ __forceinline__ unsigned short f2bf(float f) {
  unsigned int u = __float_as_uint(f);
  unsigned int r = u + 0x7FFFu + ((u >> 16) & 1u);   // round-to-nearest-even
  return (unsigned short)(r >> 16);
}

// gfx1250 async global->LDS copy, 16B per lane. VDST = LDS byte address,
// VADDR = 64-bit global address, no SADDR. Tracked by ASYNCcnt.
static __device__ __forceinline__ void async_copy_b128(unsigned lds_addr,
                                                       const void* gaddr) {
  asm volatile("global_load_async_to_lds_b128 %0, %1, off"
               :: "v"(lds_addr), "v"(gaddr) : "memory");
}
static __device__ __forceinline__ void async_wait0() {
  asm volatile("s_wait_asynccnt 0x0" ::: "memory");
}

// ---------------------------------------------------------------------------
// f32 -> bf16 conversion (weights), 4 elements / thread
// ---------------------------------------------------------------------------
__global__ __launch_bounds__(256) void cvt_f32_bf16(
    const float* __restrict__ in, unsigned short* __restrict__ out, int n4) {
  int i = blockIdx.x * 256 + threadIdx.x;
  if (i >= n4) return;
  f32x4 v = ((const f32x4*)in)[i];
  unsigned int p0 = (unsigned int)f2bf(v.x) | ((unsigned int)f2bf(v.y) << 16);
  unsigned int p1 = (unsigned int)f2bf(v.z) | ((unsigned int)f2bf(v.w) << 16);
  u32x2 o = {p0, p1};
  ((u32x2*)out)[i] = o;
}

// ---------------------------------------------------------------------------
// LayerNorm over rows of 1024 f32 -> bf16. One block (256 thr) per row.
// ---------------------------------------------------------------------------
__global__ __launch_bounds__(256) void layernorm_bf16(
    const float* __restrict__ X, const float* __restrict__ g,
    const float* __restrict__ be, unsigned short* __restrict__ Y) {
  __shared__ float ssum[8], ssq[8];
  int row = blockIdx.x, tid = threadIdx.x;
  f32x4 v = ((const f32x4*)(X + (size_t)row * 1024))[tid];
  float s = v.x + v.y + v.z + v.w;
  float q = v.x * v.x + v.y * v.y + v.z * v.z + v.w * v.w;
#pragma unroll
  for (int m = 16; m; m >>= 1) {
    s += __shfl_xor(s, m, 32);
    q += __shfl_xor(q, m, 32);
  }
  if ((tid & 31) == 0) { ssum[tid >> 5] = s; ssq[tid >> 5] = q; }
  __syncthreads();
  s = 0.f; q = 0.f;
#pragma unroll
  for (int i = 0; i < 8; ++i) { s += ssum[i]; q += ssq[i]; }
  float mu = s * (1.0f / 1024.0f);
  float var = q * (1.0f / 1024.0f) - mu * mu;
  float rs = rsqrtf(var + 1e-5f);
  f32x4 gg = ((const f32x4*)g)[tid];
  f32x4 bb = ((const f32x4*)be)[tid];
  unsigned int p0 = (unsigned int)f2bf((v.x - mu) * rs * gg.x + bb.x) |
                    ((unsigned int)f2bf((v.y - mu) * rs * gg.y + bb.y) << 16);
  unsigned int p1 = (unsigned int)f2bf((v.z - mu) * rs * gg.z + bb.z) |
                    ((unsigned int)f2bf((v.w - mu) * rs * gg.w + bb.w) << 16);
  u32x2 o = {p0, p1};
  *((u32x2*)(Y + (size_t)row * 1024) + tid) = o;
}

// ---------------------------------------------------------------------------
// Tiled bf16 WMMA GEMM: Out[M,N] = op(A[M,K] * B[K,N] (+bias) (+resid))
// Block tile 128x128, 8 waves in 2x4, wave tile 64x32 (4x2 WMMA accum),
// K-step 32. A staged via async global->LDS, double-buffered; B staged
// transposed ([n][k]) with packed b32 LDS stores, double-buffered.
// ---------------------------------------------------------------------------
template <bool BIAS, bool GELU, bool RESID, bool OUT_BF16>
__global__ __launch_bounds__(256) void gemm_bf16_wmma(
    const unsigned short* __restrict__ A,   // [M,K] bf16 row-major
    const unsigned short* __restrict__ Bm,  // [K,N] bf16 row-major
    const float* __restrict__ bias,         // [N]
    const float* __restrict__ resid,        // [M,N] f32
    void* __restrict__ Out,                 // f32 or bf16 [M,N]
    int M, int K, int N) {
  constexpr int BM = 128, BN = 128, BK = 32;
  constexpr int LDT = BK + 8;  // 40 ushorts -> 80B rows, keeps 16B alignment
  __shared__ __align__(16) unsigned short Alds[2][BM * LDT];
  __shared__ __align__(16) unsigned short Blds[2][BN * LDT];  // [n][k]

  int tid = threadIdx.x;
  int lane = tid & 31, w = tid >> 5;
  int wm = w >> 2, wn = w & 3;      // 2 (M) x 4 (N) wave grid
  int hlf = lane >> 4, l16 = lane & 15;
  int mTile = blockIdx.y * BM, nTile = blockIdx.x * BN;

  // Staging assignments
  int arow = tid >> 1;              // 128 rows, 2 threads/row
  int aseg0 = (tid & 1) * 2;        // each thread: segments aseg0, aseg0+1 (16B)
  int bk = (tid >> 4) * 2;          // 16 k-pairs
  int bn0 = (tid & 15) * 8;         // 16 col-octets

  v8f acc[4][2] = {};

  auto stage = [&](int buf, int k0) {
    // A tile: 128x32 bf16, 32B/thread via two async b128 copies.
#pragma unroll
    for (int j = 0; j < 2; ++j) {
      int seg = aseg0 + j;
      unsigned lds = (unsigned)(size_t)&Alds[buf][arow * LDT + seg * 8];
      const void* gp = (const void*)(A + (size_t)(mTile + arow) * K + k0 + seg * 8);
      async_copy_b128(lds, gp);
    }
    // B tile transposed: load two K-rows, store 8 packed b32 (k,k+1 pairs).
    u32x4 r0 = *(const u32x4*)(Bm + (size_t)(k0 + bk) * N + nTile + bn0);
    u32x4 r1 = *(const u32x4*)(Bm + (size_t)(k0 + bk + 1) * N + nTile + bn0);
    const unsigned short* p0 = (const unsigned short*)&r0;
    const unsigned short* p1 = (const unsigned short*)&r1;
#pragma unroll
    for (int i = 0; i < 8; ++i) {
      unsigned int pk = (unsigned int)p0[i] | ((unsigned int)p1[i] << 16);
      *(unsigned int*)(&Blds[buf][(bn0 + i) * LDT + bk]) = pk;
    }
  };

  stage(0, 0);
  async_wait0();
  __syncthreads();

  int nk = K / BK;
  for (int kt = 0; kt < nk; ++kt) {
    int cur = kt & 1;
    if (kt + 1 < nk) stage(cur ^ 1, (kt + 1) * BK);

    v16bf af[4], bfv[2];
#pragma unroll
    for (int tm = 0; tm < 4; ++tm) {
      const unsigned short* base = &Alds[cur][(wm * 64 + tm * 16 + l16) * LDT];
      af[tm] = mkfrag(*(const u32x4*)(base + 8 * hlf),
                      *(const u32x4*)(base + 16 + 8 * hlf));
    }
#pragma unroll
    for (int tn = 0; tn < 2; ++tn) {
      const unsigned short* base = &Blds[cur][(wn * 32 + tn * 16 + l16) * LDT];
      bfv[tn] = mkfrag(*(const u32x4*)(base + 8 * hlf),
                       *(const u32x4*)(base + 16 + 8 * hlf));
    }
#pragma unroll
    for (int tm = 0; tm < 4; ++tm)
#pragma unroll
      for (int tn = 0; tn < 2; ++tn)
        acc[tm][tn] = wmma_bf16(af[tm], bfv[tn], acc[tm][tn]);

    async_wait0();
    __syncthreads();
  }

  // Epilogue: D layout is row = r + 8*hlf, col = l16.
#pragma unroll
  for (int tm = 0; tm < 4; ++tm) {
    int rbase = mTile + wm * 64 + tm * 16 + 8 * hlf;
#pragma unroll
    for (int tn = 0; tn < 2; ++tn) {
      int col = nTile + wn * 32 + tn * 16 + l16;
      float bc = BIAS ? bias[col] : 0.0f;
#pragma unroll
      for (int r = 0; r < 8; ++r) {
        float v = acc[tm][tn][r] + bc;
        if (GELU) v = 0.5f * v * (1.0f + erff(v * 0.70710678118654752f));
        size_t idx = (size_t)(rbase + r) * N + col;
        if (RESID) v += resid[idx];
        if (OUT_BF16)
          ((unsigned short*)Out)[idx] = f2bf(v);
        else
          ((float*)Out)[idx] = v;
      }
    }
  }
}

// ---------------------------------------------------------------------------
// Flash attention. qkv: [B*2048, 3072] bf16 with col = s*1024 + h*64 + d.
// Block = (b,h, 64-row q-block); 4 waves, each owns a 16-row Q tile.
// ---------------------------------------------------------------------------
__global__ __launch_bounds__(128) void attention_wmma(
    const unsigned short* __restrict__ qkv, unsigned short* __restrict__ O) {
  constexpr int NQ = 2048;
  constexpr int LDV = 40;  // VldsT row stride (ushorts) -> 80B, 16B-aligned
  __shared__ __align__(16) unsigned short VldsT[64 * LDV];  // [d][k]
  __shared__ __align__(16) unsigned short Plds[4][16 * 32]; // per-wave P tile

  int lane = threadIdx.x & 31, w = threadIdx.x >> 5;
  int hlf = lane >> 4, l16 = lane & 15;
  int qb = blockIdx.x & 31;
  int bh = blockIdx.x >> 5;
  int b = bh >> 4, h = bh & 15;
  const size_t RS = 3072;
  const unsigned short* base = qkv + (size_t)(b * NQ) * RS;

  // Q fragments (A layout), this lane's row = qb*64 + w*16 + l16.
  v16bf aq[2];
  {
    const unsigned short* qp = base + (size_t)(qb * 64 + w * 16 + l16) * RS + h * 64;
#pragma unroll
    for (int t = 0; t < 2; ++t)
      aq[t] = mkfrag(*(const u32x4*)(qp + 32 * t + 8 * hlf),
                     *(const u32x4*)(qp + 32 * t + 16 + 8 * hlf));
  }

  float mrow[8], lrow[8];
  v8f accO[4] = {};
#pragma unroll
  for (int r = 0; r < 8; ++r) { mrow[r] = -1e30f; lrow[r] = 0.f; }

  for (int jb = 0; jb < NQ / 32; ++jb) {
    // Cooperative V staging, transposed: VldsT[d][k_local].
    {
      int k = threadIdx.x >> 2;          // 0..31 key in block
      int d0 = (threadIdx.x & 3) * 16;   // 16-wide d chunk
      const unsigned short* vp = base + (size_t)(jb * 32 + k) * RS + 2048 + h * 64 + d0;
      u32x4 u0 = *(const u32x4*)(vp);
      u32x4 u1 = *(const u32x4*)(vp + 8);
      const unsigned short* p0 = (const unsigned short*)&u0;
      const unsigned short* p1 = (const unsigned short*)&u1;
#pragma unroll
      for (int i = 0; i < 8; ++i) {
        VldsT[(d0 + i) * LDV + k] = p0[i];
        VldsT[(d0 + 8 + i) * LDV + k] = p1[i];
      }
    }

    // S = Q * K^T for 32 keys: two 16x16 tiles, K fragments from global.
    v8f sacc[2];
#pragma unroll
    for (int jt = 0; jt < 2; ++jt) {
      v8f z = {};
      const unsigned short* kp =
          base + (size_t)(jb * 32 + jt * 16 + l16) * RS + 1024 + h * 64;
#pragma unroll
      for (int t = 0; t < 2; ++t) {
        v16bf bk = mkfrag(*(const u32x4*)(kp + 32 * t + 8 * hlf),
                          *(const u32x4*)(kp + 32 * t + 16 + 8 * hlf));
        z = wmma_bf16(aq[t], bk, z);
      }
      sacc[jt] = z;
    }

    // Online softmax (row = r + 8*hlf lives in this lane's 16-lane group).
    float alpha[8];
#pragma unroll
    for (int r = 0; r < 8; ++r) {
      float s0 = sacc[0][r] * 0.125f;   // 1/sqrt(64)
      float s1 = sacc[1][r] * 0.125f;
      float mx = fmaxf(s0, s1);
#pragma unroll
      for (int m = 1; m <= 8; m <<= 1) mx = fmaxf(mx, __shfl_xor(mx, m, 32));
      float mn = fmaxf(mrow[r], mx);
      float a = __expf(mrow[r] - mn);
      float p0 = __expf(s0 - mn);
      float p1 = __expf(s1 - mn);
      float ps = p0 + p1;
#pragma unroll
      for (int m = 1; m <= 8; m <<= 1) ps += __shfl_xor(ps, m, 32);
      lrow[r] = lrow[r] * a + ps;
      mrow[r] = mn;
      alpha[r] = a;
      int prow = r + 8 * hlf;
      Plds[w][prow * 32 + l16] = f2bf(p0);
      Plds[w][prow * 32 + 16 + l16] = f2bf(p1);
    }
#pragma unroll
    for (int nt = 0; nt < 4; ++nt)
#pragma unroll
      for (int r = 0; r < 8; ++r) accO[nt][r] *= alpha[r];

    __syncthreads();  // V staged, P visible

    // P as A-fragment from LDS, V B-fragments from transposed LDS; O += P*V.
    v16bf pf;
    {
      const unsigned short* pp = &Plds[w][l16 * 32];
      pf = mkfrag(*(const u32x4*)(pp + 8 * hlf),
                  *(const u32x4*)(pp + 16 + 8 * hlf));
    }
#pragma unroll
    for (int nt = 0; nt < 4; ++nt) {
      const unsigned short* vp = VldsT + (nt * 16 + l16) * LDV;
      v16bf vf = mkfrag(*(const u32x4*)(vp + 8 * hlf),
                        *(const u32x4*)(vp + 16 + 8 * hlf));
      accO[nt] = wmma_bf16(pf, vf, accO[nt]);
    }
    __syncthreads();  // protect LDS tiles before next iteration
  }

  // Normalize and write O (bf16) at [row, h*64 + d].
#pragma unroll
  for (int r = 0; r < 8; ++r) {
    float inv = 1.0f / lrow[r];
    int gr = qb * 64 + w * 16 + r + 8 * hlf;
    size_t ob = (size_t)(b * NQ + gr) * 1024 + h * 64;
#pragma unroll
    for (int nt = 0; nt < 4; ++nt)
      O[ob + nt * 16 + l16] = f2bf(accO[nt][r] * inv);
  }
}

// ---------------------------------------------------------------------------
// Host-side orchestration
// ---------------------------------------------------------------------------
extern "C" void kernel_launch(void* const* d_in, const int* in_sizes, int n_in,
                              void* d_out, int out_size, void* d_ws, size_t ws_size,
                              hipStream_t stream) {
  (void)in_sizes; (void)n_in; (void)out_size; (void)ws_size;
  const float* x     = (const float*)d_in[0];
  const float* ln1w  = (const float*)d_in[1];
  const float* ln1b  = (const float*)d_in[2];
  const float* wqkv  = (const float*)d_in[3];
  const float* wproj = (const float*)d_in[4];
  const float* ln2w  = (const float*)d_in[5];
  const float* ln2b  = (const float*)d_in[6];
  const float* wfc1  = (const float*)d_in[7];
  const float* bfc1  = (const float*)d_in[8];
  const float* wfc2  = (const float*)d_in[9];
  const float* bfc2  = (const float*)d_in[10];

  const int M = 4096;          // B*N tokens
  char* ws = (char*)d_ws;
  unsigned short* wqkv_bf  = (unsigned short*)(ws + 0);          //  6.0 MB
  unsigned short* wproj_bf = (unsigned short*)(ws + 6291456);    //  2.0 MB
  unsigned short* wfc1_bf  = (unsigned short*)(ws + 8388608);    //  8.0 MB
  unsigned short* wfc2_bf  = (unsigned short*)(ws + 16777216);   //  8.0 MB
  unsigned short* qkv_bf   = (unsigned short*)(ws + 25165824);   // 24.0 MB
  unsigned short* o_bf     = (unsigned short*)(ws + 50331648);   //  8.0 MB
  float*          x1       = (float*)(ws + 58720256);            // 16.0 MB
  unsigned short* h_bf     = (unsigned short*)(ws + 75497472);   //  8.0 MB
  unsigned short* g_bf     = qkv_bf;  // alias: qkv dead once GELU acts computed

  // Weight conversions f32 -> bf16
  cvt_f32_bf16<<<(1024 * 3072 / 4) / 256, 256, 0, stream>>>(wqkv, wqkv_bf, 1024 * 3072 / 4);
  cvt_f32_bf16<<<(1024 * 1024 / 4) / 256, 256, 0, stream>>>(wproj, wproj_bf, 1024 * 1024 / 4);
  cvt_f32_bf16<<<(1024 * 4096 / 4) / 256, 256, 0, stream>>>(wfc1, wfc1_bf, 1024 * 4096 / 4);
  cvt_f32_bf16<<<(4096 * 1024 / 4) / 256, 256, 0, stream>>>(wfc2, wfc2_bf, 4096 * 1024 / 4);

  // h = LN1(x) in bf16
  layernorm_bf16<<<M, 256, 0, stream>>>(x, ln1w, ln1b, h_bf);

  // qkv = h @ w_qkv   [4096,3072] bf16
  gemm_bf16_wmma<false, false, false, true>
      <<<dim3(3072 / 128, M / 128), 256, 0, stream>>>(
          h_bf, wqkv_bf, nullptr, nullptr, qkv_bf, M, 1024, 3072);

  // o = softmax(QK^T/sqrt(Dh)) V   (flash, per (b,h,qblock))
  attention_wmma<<<1024, 128, 0, stream>>>(qkv_bf, o_bf);

  // x1 = x + o @ w_proj   f32
  gemm_bf16_wmma<false, false, true, false>
      <<<dim3(1024 / 128, M / 128), 256, 0, stream>>>(
          o_bf, wproj_bf, nullptr, x, x1, M, 1024, 1024);

  // h = LN2(x1) bf16
  layernorm_bf16<<<M, 256, 0, stream>>>(x1, ln2w, ln2b, h_bf);

  // g = gelu(h @ w_fc1 + b_fc1)  bf16
  gemm_bf16_wmma<true, true, false, true>
      <<<dim3(4096 / 128, M / 128), 256, 0, stream>>>(
          h_bf, wfc1_bf, bfc1, nullptr, g_bf, M, 1024, 4096);

  // out = x1 + g @ w_fc2 + b_fc2  f32 -> d_out
  gemm_bf16_wmma<true, false, true, false>
      <<<dim3(1024 / 128, M / 128), 256, 0, stream>>>(
          g_bf, wfc2_bf, bfc2, x1, (float*)d_out, M, 4096, 1024);
}
